// TripletLoss_26319559590784
// MI455X (gfx1250) — matrix operand order repaired and checked by reading, not compiled, not run on hardware
//
#include <hip/hip_runtime.h>

typedef __attribute__((ext_vector_type(16))) _Float16 v16h;
typedef __attribute__((ext_vector_type(8)))  float    v8f;

#define TL_D      128
#define TL_MARGIN 0.05f
#define TL_NCLS   4

struct TlMeta {
    int cnt[TL_NCLS];   // class histogram
    int off[TL_NCLS];   // padded segment starts (multiples of 16)
    int mp;             // padded column count
};

// Single-instruction min/max accumulate (avoids llvm.minnum canonicalization
// traffic: no v_max v,v,v quieting ops).
__device__ __forceinline__ void tl_accmin(float& a, float b) {
    asm("v_min_num_f32 %0, %0, %1" : "+v"(a) : "v"(b));
}
__device__ __forceinline__ void tl_accmax(float& a, float b) {
    asm("v_max_num_f32 %0, %0, %1" : "+v"(a) : "v"(b));
}

// ---------------------------------------------------------------------------
// Kernel A: single block. Histogram the 4 classes, build padded class-sorted
// column permutation (perm: padded position -> original row, pos: original
// row -> padded position) and zero the accumulators. Pad slots duplicate the
// first member of their class (duplicates are min/max-neutral).
// ---------------------------------------------------------------------------
__global__ __launch_bounds__(256) void tl_classify(const int* __restrict__ lbl, int n,
                                                   int* __restrict__ pos,
                                                   int* __restrict__ perm,
                                                   TlMeta* __restrict__ meta,
                                                   float* __restrict__ acc)
{
    __shared__ int h[TL_NCLS];
    __shared__ int sc[TL_NCLS];
    __shared__ int soff[TL_NCLS];
    __shared__ int smp;
    const int tid = threadIdx.x;

    if (tid < TL_NCLS) h[tid] = 0;
    if (tid == 0) { acc[0] = 0.0f; acc[1] = 0.0f; }
    __syncthreads();

    for (int i = tid; i < n; i += 256) atomicAdd(&h[lbl[i] & 3], 1);
    __syncthreads();

    if (tid == 0) {
        int o = 0;
        for (int c = 0; c < TL_NCLS; ++c) {
            soff[c]      = o;
            meta->cnt[c] = h[c];
            meta->off[c] = o;
            o += (h[c] + 15) & ~15;           // 16-align each segment
        }
        smp      = o;
        meta->mp = o;
    }
    __syncthreads();

    if (tid < TL_NCLS) sc[tid] = soff[tid];
    __syncthreads();

    for (int i = tid; i < n; i += 256) {
        const int p = atomicAdd(&sc[lbl[i] & 3], 1);
        perm[p] = i;
        pos[i]  = p;
    }
    __syncthreads();

    // pad slots: duplicate the first member of the class
    if (tid < TL_NCLS) {
        const int c = tid;
        const int s = soff[c] + h[c];
        const int e = (c + 1 < TL_NCLS) ? soff[c + 1] : smp;
        for (int p = s; p < e; ++p) perm[p] = perm[soff[c]];
    }
}

// ---------------------------------------------------------------------------
// Kernel B: row L2-normalize f32 -> f16, scattered into class-sorted order.
// ---------------------------------------------------------------------------
__global__ __launch_bounds__(TL_D) void tl_normalize(const float* __restrict__ x,
                                                     const int* __restrict__ pos,
                                                     _Float16* __restrict__ xp)
{
    const int row = blockIdx.x;
    const int tid = threadIdx.x;

    const float v = x[(size_t)row * TL_D + tid];
    float ss = v * v;
#pragma unroll
    for (int m = 16; m > 0; m >>= 1) ss += __shfl_xor(ss, m, 32);

    __shared__ float sred[4];
    __shared__ float rn_s;
    if ((tid & 31) == 0) sred[tid >> 5] = ss;
    __syncthreads();
    if (tid == 0) rn_s = rsqrtf(sred[0] + sred[1] + sred[2] + sred[3]);
    __syncthreads();

    xp[(size_t)pos[row] * TL_D + tid] = (_Float16)(v * rn_s);
}

// ---------------------------------------------------------------------------
// Kernel C: fill pad rows with a copy of the first member of their class.
// ---------------------------------------------------------------------------
__global__ __launch_bounds__(TL_D) void tl_pad(const TlMeta* __restrict__ meta,
                                               _Float16* __restrict__ xp)
{
    const int c   = blockIdx.x;
    const int tid = threadIdx.x;
    const int cnt = meta->cnt[c];
    const int off = meta->off[c];
    const int s   = off + cnt;
    const int e   = off + ((cnt + 15) & ~15);
    if (s >= e) return;
    const _Float16 v = xp[(size_t)off * TL_D + tid];
    for (int p = s; p < e; ++p) xp[(size_t)p * TL_D + tid] = v;
}

// ---------------------------------------------------------------------------
// Mining helpers
// ---------------------------------------------------------------------------
// Load one B tile's four 32x16 f16 fragments (K = 128) given the precomputed lane base pointer.
__device__ __forceinline__ void tl_load_bfrag(const _Float16* __restrict__ brow,
                                              int sel, v16h* b)
{
#pragma unroll
    for (int k = 0; k < 4; ++k) {
        const int kb = 32 * k + 8 * sel;
        reinterpret_cast<float4*>(&b[k])[0] =
            *reinterpret_cast<const float4*>(brow + kb);
        reinterpret_cast<float4*>(&b[k])[1] =
            *reinterpret_cast<const float4*>(brow + kb + 16);
    }
}

__device__ __forceinline__ void tl_tile(const v16h* afr, const v16h* b,
                                        float* amin, float* amax)
{
    v8f cc = {0.f, 0.f, 0.f, 0.f, 0.f, 0.f, 0.f, 0.f};
#pragma unroll
    for (int k = 0; k < 4; ++k)
        cc = __builtin_amdgcn_wmma_f32_16x16x32_f16(false, afr[k],
                                                    false, b[k],
                                                    (short)0, cc,
                                                    false, false);
#pragma unroll
    for (int r = 0; r < 8; ++r) {
        tl_accmin(amin[r], cc[r]);
        tl_accmax(amax[r], cc[r]);
    }
}

// ---------------------------------------------------------------------------
// Kernel D: batch-hard miner over class-sorted columns. One block = 16 anchor
// rows; 8 waves split each class segment's tiles. Steady-state loop is
// unrolled by 2 with FIXED buffer roles (no conditional buffer selection ->
// no phi register copies, no in-loop branches): tile(b0); reload b0;
// tile(b1); reload b1. Fold into posmin/negmax once per class.
// ---------------------------------------------------------------------------
__global__ __launch_bounds__(256) void tl_mine(const _Float16* __restrict__ xp,
                                               const int* __restrict__ pos,
                                               const int* __restrict__ lbl,
                                               const TlMeta* __restrict__ meta,
                                               float* __restrict__ acc,
                                               int n)
{
    const int tid  = threadIdx.x;
    const int lane = tid & 31;
    const int wave = tid >> 5;      // 0..7
    const int sel  = lane >> 4;     // 16-lane half
    const int m    = lane & 15;
    const int i0   = blockIdx.x * 16;

    // per-wave tile stride in halves: 8 tiles * 16 rows * TL_D
    const size_t TSTRIDE = (size_t)8 * 16 * TL_D;

    // segment tile boundaries (uniform -> scalar loads)
    int segs[TL_NCLS + 1];
#pragma unroll
    for (int c = 0; c < TL_NCLS; ++c) segs[c] = meta->off[c];
    segs[TL_NCLS] = meta->mp;

    // anchor labels for the 8 C rows this half-wave covers
    const int rbase = i0 + 8 * sel;
    int albl[8];
#pragma unroll
    for (int r = 0; r < 8; ++r) albl[r] = lbl[rbase + r] & 3;

    // A fragments (16x32 f16, ISA A-layout) gathered via pos[] (anchors keep
    // original order; only columns are permuted).
    v16h afr[4];
    {
        const _Float16* arow = xp + (size_t)pos[i0 + m] * TL_D;
#pragma unroll
        for (int k = 0; k < 4; ++k) {
            const int kb = 32 * k + 8 * sel;
            reinterpret_cast<float4*>(&afr[k])[0] =
                *reinterpret_cast<const float4*>(arow + kb);
            reinterpret_cast<float4*>(&afr[k])[1] =
                *reinterpret_cast<const float4*>(arow + kb + 16);
        }
    }

    float posmin[8], negmax[8];   // hardest-pos = min dot, hardest-neg = max dot
#pragma unroll
    for (int r = 0; r < 8; ++r) { posmin[r] = 1e30f; negmax[r] = -1e30f; }

    for (int c = 0; c < TL_NCLS; ++c) {
        const int t0 = segs[c] >> 4;
        const int t1 = segs[c + 1] >> 4;

        float amin[8], amax[8];
#pragma unroll
        for (int r = 0; r < 8; ++r) { amin[r] = 1e30f; amax[r] = -1e30f; }

        const int first = t0 + wave;
        if (first < t1) {
            const int cnt = (t1 - first + 7) >> 3;   // tiles for this wave
            const _Float16* base = xp + (size_t)(first * 16 + m) * TL_D;

            v16h b0[4], b1[4];
            tl_load_bfrag(base, sel, b0);
            if (cnt > 1) tl_load_bfrag(base + TSTRIDE, sel, b1);

            int i = 0;
            // steady state: consume 2, prefetch 2 (fixed buffer roles)
            for (; i + 3 < cnt; i += 2) {
                tl_tile(afr, b0, amin, amax);
                tl_load_bfrag(base + (size_t)(i + 2) * TSTRIDE, sel, b0);
                tl_tile(afr, b1, amin, amax);
                tl_load_bfrag(base + (size_t)(i + 3) * TSTRIDE, sel, b1);
            }
            // tail: 1..3 tiles remain (b0 = tile i, b1 = tile i+1 if present)
            const int rem = cnt - i;
            if (rem == 3) {
                tl_tile(afr, b0, amin, amax);
                tl_load_bfrag(base + (size_t)(i + 2) * TSTRIDE, sel, b0);
                tl_tile(afr, b1, amin, amax);
                tl_tile(afr, b0, amin, amax);
            } else if (rem == 2) {
                tl_tile(afr, b0, amin, amax);
                tl_tile(afr, b1, amin, amax);
            } else {
                tl_tile(afr, b0, amin, amax);
            }
        }

        // fold this class segment (4x per kernel, negligible)
#pragma unroll
        for (int r = 0; r < 8; ++r) {
            const bool own = (albl[r] == c);
            posmin[r] = own ? fminf(posmin[r], amin[r]) : posmin[r];
            negmax[r] = own ? negmax[r] : fmaxf(negmax[r], amax[r]);
        }
    }

    // reduce across the 16 lanes of each half (xor masks <16 stay in-half)
#pragma unroll
    for (int r = 0; r < 8; ++r) {
#pragma unroll
        for (int msk = 8; msk > 0; msk >>= 1) {
            posmin[r] = fminf(posmin[r], __shfl_xor(posmin[r], msk, 32));
            negmax[r] = fmaxf(negmax[r], __shfl_xor(negmax[r], msk, 32));
        }
    }

    __shared__ float s_pos[16][8];
    __shared__ float s_neg[16][8];
    if (m == 0) {
#pragma unroll
        for (int r = 0; r < 8; ++r) {
            s_pos[8 * sel + r][wave] = posmin[r];
            s_neg[8 * sel + r][wave] = negmax[r];
        }
    }
    __syncthreads();

    if (tid < 32) {
        const int row = tid & 15;
        float pm = 1e30f, nm = -1e30f;
#pragma unroll
        for (int w = 0; w < 8; ++w) {
            pm = fminf(pm, s_pos[row][w]);
            nm = fmaxf(nm, s_neg[row][w]);
        }
        const int  lb    = lbl[i0 + row] & 3;
        const int  ccnt  = meta->cnt[lb];
        const bool valid = (ccnt >= 2) && (n - ccnt >= 1);
        const float hp = sqrtf(fmaxf(2.0f - 2.0f * pm, 0.0f)); // max pos dist
        const float hn = sqrtf(fmaxf(2.0f - 2.0f * nm, 0.0f)); // min neg dist
        float la  = valid ? fmaxf(hp - hn + TL_MARGIN, 0.0f) : 0.0f;
        float cnt = (la > 0.0f) ? 1.0f : 0.0f;
#pragma unroll
        for (int msk = 8; msk > 0; msk >>= 1) {
            la  += __shfl_xor(la,  msk, 32);
            cnt += __shfl_xor(cnt, msk, 32);
        }
        if (tid == 0) {
            atomicAdd(&acc[0], la);
            atomicAdd(&acc[1], cnt);
        }
    }
}

// ---------------------------------------------------------------------------
// Kernel E: loss = sum / max(nnz, 1)
// ---------------------------------------------------------------------------
__global__ void tl_finalize(const float* __restrict__ acc, float* __restrict__ out)
{
    out[0] = acc[0] / fmaxf(acc[1], 1.0f);
}

extern "C" void kernel_launch(void* const* d_in, const int* in_sizes, int n_in,
                              void* d_out, int out_size, void* d_ws, size_t ws_size,
                              hipStream_t stream)
{
    const float* x   = (const float*)d_in[0];   // [N, 128] f32
    const int*   lbl = (const int*)d_in[1];     // [N] labels (0..3)
    const int    n   = in_sizes[1];             // 8192
    const int    mpmax = n + 16 * TL_NCLS;      // padded column capacity

    char* w = (char*)d_ws;
    _Float16* xp   = (_Float16*)w;  w += (size_t)mpmax * TL_D * sizeof(_Float16);
    int*      pos  = (int*)w;       w += (size_t)n * sizeof(int);
    int*      perm = (int*)w;       w += (size_t)mpmax * sizeof(int);
    TlMeta*   meta = (TlMeta*)w;    w += sizeof(TlMeta);
    float*    acc  = (float*)w;

    tl_classify<<<1, 256, 0, stream>>>(lbl, n, pos, perm, meta, acc);
    tl_normalize<<<n, TL_D, 0, stream>>>(x, pos, xp);
    tl_pad<<<TL_NCLS, TL_D, 0, stream>>>(meta, xp);
    tl_mine<<<n / 16, 256, 0, stream>>>(xp, pos, lbl, meta, acc, n);
    tl_finalize<<<1, 1, 0, stream>>>(acc, (float*)d_out);
}
